// CausalSelfAttention_59407987638752
// MI455X (gfx1250) — matrix operand order repaired and checked by reading, not compiled, not run on hardware
//
#include <hip/hip_runtime.h>

typedef __attribute__((ext_vector_type(16))) _Float16 v16h;
typedef __attribute__((ext_vector_type(8)))  _Float16 v8h;
typedef __attribute__((ext_vector_type(8)))  float    v8f;
typedef __attribute__((ext_vector_type(4)))  float    v4f;

#define T_DIM 4096
#define C_DIM 128
#define B_DIM 4

static __device__ inline v8f wmma_f16(v16h a, v16h b, v8f c) {
  // D = A(16x32 f16) * B(32x16 f16) + C(16x16 f32)
  return __builtin_amdgcn_wmma_f32_16x16x32_f16(false, a, false, b, (short)0, c,
                                                false, false);
}

static __device__ inline v16h cat8(v8h lo, v8h hi) {
  return __builtin_shufflevector(lo, hi, 0, 1, 2, 3, 4, 5, 6, 7, 8, 9, 10, 11,
                                 12, 13, 14, 15);
}

static __device__ inline float bcastf(float v, int srcLane) {
  return __int_as_float(
      __builtin_amdgcn_ds_bpermute(srcLane << 2, __float_as_int(v)));
}

// ---------------------------------------------------------------------------
// Kernel 0: one-shot f32 -> f16 conversion of Wq|Wk|Wv into workspace.
// wf layout: [w][d][c], 3*C*C halves (96 KB, L2-resident).
// ---------------------------------------------------------------------------
__global__ __launch_bounds__(256) void cvtw_kernel(
    const float* __restrict__ Wq, const float* __restrict__ Wk,
    const float* __restrict__ Wv, _Float16* __restrict__ wf) {
  const int i = (blockIdx.x * 256 + threadIdx.x) * 8;  // 0 .. 3*C*C-1, 8/thrd
  const int m = C_DIM * C_DIM;
  const float* src = (i < m) ? (Wq + i) : (i < 2 * m) ? (Wk + i - m)
                                                      : (Wv + i - 2 * m);
  v8h o;
#pragma unroll
  for (int e = 0; e < 8; ++e) o[e] = (_Float16)src[e];
  *(v8h*)(wf + i) = o;
}

// ---------------------------------------------------------------------------
// Kernel 1: QKV projection.  y = x @ W^T  (W row-major [d][c], so B-fragment
// lanes (n = d) read W rows contiguously).  One wave per 16-row tile.
// q,k stored f16 row-major [b*T + t][C]; v stored transposed [b][C][T].
// ---------------------------------------------------------------------------
__global__ __launch_bounds__(128) void qkv_kernel(
    const float* __restrict__ x, const _Float16* __restrict__ wf,
    _Float16* __restrict__ qh, _Float16* __restrict__ kh,
    _Float16* __restrict__ vth) {
  const int wave = threadIdx.x >> 5;
  const int tile = blockIdx.x * 4 + wave;      // 0 .. B*T/16-1
  const int row0 = tile * 16;                  // flat row index b*T + t
  const int b    = row0 / T_DIM;
  const int t0   = row0 % T_DIM;
  const int l    = threadIdx.x & 31;
  const int r    = l & 15;
  const int hi   = l >> 4;

  // A-fragments of x: lane row = row0 + r, K pattern {hi*8..+7, 16+hi*8..+7}
  v16h xa[4];
  const float* xrow = x + (size_t)(row0 + r) * C_DIM;
#pragma unroll
  for (int kc = 0; kc < 4; ++kc) {
    const int c0 = kc * 32 + hi * 8;
#pragma unroll
    for (int e = 0; e < 8; ++e) xa[kc][e] = (_Float16)xrow[c0 + e];
#pragma unroll
    for (int e = 0; e < 8; ++e) xa[kc][8 + e] = (_Float16)xrow[c0 + 16 + e];
  }

#pragma unroll
  for (int w = 0; w < 3; ++w) {
#pragma unroll 2
    for (int nt = 0; nt < 8; ++nt) {
      const int d = nt * 16 + r;                     // B-fragment: lane n = d
      const _Float16* wrow = wf + (size_t)w * C_DIM * C_DIM + (size_t)d * C_DIM;
      v8f acc = {0.f, 0.f, 0.f, 0.f, 0.f, 0.f, 0.f, 0.f};
#pragma unroll
      for (int kc = 0; kc < 4; ++kc) {
        const v16h wb = cat8(*(const v8h*)(wrow + kc * 32 + hi * 8),
                             *(const v8h*)(wrow + kc * 32 + hi * 8 + 16));
        acc = wmma_f16(xa[kc], wb, acc);
      }
      // D layout: vgpr v -> (t_local = v + 8*hi, d_local = r)
      if (w < 2) {
        _Float16* dst = (w == 0) ? qh : kh;
#pragma unroll
        for (int v = 0; v < 8; ++v) {
          const int tl = v + 8 * hi;
          dst[(size_t)(row0 + tl) * C_DIM + nt * 16 + r] = (_Float16)acc[v];
        }
      } else {
        v8h pk;
#pragma unroll
        for (int v = 0; v < 8; ++v) pk[v] = (_Float16)acc[v];
        const int dcol = nt * 16 + r;                // 8 consecutive t values
        _Float16* dst = vth + (size_t)b * C_DIM * T_DIM +
                        (size_t)dcol * T_DIM + (t0 + 8 * hi);
        *(v8h*)dst = pk;
      }
    }
  }
}

// ---------------------------------------------------------------------------
// One 32-key flash-attention block.  S^T = K * Q^T so the probability tiles
// land directly in the A-fragment layout for P @ V (no cross-lane movement).
// MASKED=true only for the single diagonal-straddling block per wave.
// ---------------------------------------------------------------------------
template <bool MASKED>
static __device__ __forceinline__ void attn_block(
    int s0, int t_glob, int l, int r, int hi, const _Float16* __restrict__ kb,
    const v16h* qbf, const _Float16* __restrict__ vb, float& m_i, float& l_i,
    v8f* acc) {
  const float sc = 1.4426950408889634f * 0.08838834764831845f;  // log2e/sqrt(C)

  // ---- S^T tiles: st0 covers keys s0..s0+15, st1 covers s0+16..s0+31 ----
  v8f st0 = {0.f, 0.f, 0.f, 0.f, 0.f, 0.f, 0.f, 0.f};
  v8f st1 = {0.f, 0.f, 0.f, 0.f, 0.f, 0.f, 0.f, 0.f};
#pragma unroll
  for (int kc = 0; kc < 4; ++kc) {
    const _Float16* kr0 = kb + (size_t)(s0 + r) * C_DIM + kc * 32 + hi * 8;
    const _Float16* kr1 = kr0 + 16 * C_DIM;
    const v16h ka0 = cat8(*(const v8h*)(kr0), *(const v8h*)(kr0 + 16));
    const v16h ka1 = cat8(*(const v8h*)(kr1), *(const v8h*)(kr1 + 16));
    st0 = wmma_f16(ka0, qbf[kc], st0);
    st1 = wmma_f16(ka1, qbf[kc], st1);
  }

  // ---- scale (+ causal mask); element (v): s = s0 + v + 8*hi (+16 st1) ----
  float bm = -__builtin_inff();
#pragma unroll
  for (int v = 0; v < 8; ++v) {
    float a0 = st0[v] * sc;
    float a1 = st1[v] * sc;
    if (MASKED) {
      const int s = s0 + v + 8 * hi;
      if (s > t_glob) a0 = -__builtin_inff();
      if (s + 16 > t_glob) a1 = -__builtin_inff();
    }
    st0[v] = a0;
    st1[v] = a1;
    bm = fmaxf(bm, fmaxf(a0, a1));
  }
  bm = fmaxf(bm, bcastf(bm, l ^ 16));            // other 8 s-values: lane^16
  const float m_new = fmaxf(m_i, bm);
  const float corr = exp2f(m_i - m_new);

  float rs = 0.f;
#pragma unroll
  for (int v = 0; v < 8; ++v) {
    const float p0 = exp2f(st0[v] - m_new);
    const float p1 = exp2f(st1[v] - m_new);
    st0[v] = p0;
    st1[v] = p1;
    rs += p0 + p1;
  }
  rs += bcastf(rs, l ^ 16);
  l_i = l_i * corr + rs;
  m_i = m_new;

  // ---- P A-fragment: D-layout of S^T == A-layout of P (same lane!) ----
  v16h pa;
#pragma unroll
  for (int v = 0; v < 8; ++v) {
    pa[v]     = (_Float16)st0[v];
    pa[8 + v] = (_Float16)st1[v];
  }

  // ---- rescale O accumulators: cv[v] = corr for row t = v + 8*hi ----
  float cv[8];
#pragma unroll
  for (int v = 0; v < 8; ++v) cv[v] = bcastf(corr, v + 8 * hi);
#pragma unroll
  for (int dt = 0; dt < 8; ++dt) {
#pragma unroll
    for (int v = 0; v < 8; ++v) acc[dt][v] *= cv[v];
  }

  // ---- O += P @ V: B-frag from v^T, lane n = d, K = s0 + {hi*8..+7, 16+hi*8..+7} ----
#pragma unroll
  for (int dt = 0; dt < 8; ++dt) {
    const _Float16* vp = vb + (size_t)(dt * 16 + r) * T_DIM + s0 + hi * 8;
    const v16h vfrag = cat8(*(const v8h*)(vp), *(const v8h*)(vp + 16));
    acc[dt] = wmma_f16(pa, vfrag, acc[dt]);
  }
}

// ---------------------------------------------------------------------------
// Kernel 2: causal flash attention.  One wave per 16 query rows; key blocks
// of 32.  Main loop is unmasked; exactly one diagonal block takes the mask.
// ---------------------------------------------------------------------------
__global__ __launch_bounds__(128) void attn_kernel(
    const _Float16* __restrict__ qh, const _Float16* __restrict__ kh,
    const _Float16* __restrict__ vth, float* __restrict__ out) {
  const int wave = threadIdx.x >> 5;
  const int tile = blockIdx.x * 4 + wave;      // 0 .. B*T/16-1
  const int b    = tile / (T_DIM / 16);
  const int qt   = (tile % (T_DIM / 16)) * 16; // query base row within batch
  const int l    = threadIdx.x & 31;
  const int r    = l & 15;
  const int hi   = l >> 4;

  const _Float16* qb = qh + (size_t)b * T_DIM * C_DIM;
  const _Float16* kb = kh + (size_t)b * T_DIM * C_DIM;
  const _Float16* vb = vth + (size_t)b * C_DIM * T_DIM;

  // Q^T B-fragments: lane n = t = qt + r, K = d = kc*32 + {hi*8..+7, 16+hi*8..+7}
  v16h qbf[4];
  const _Float16* qrow = qb + (size_t)(qt + r) * C_DIM;
#pragma unroll
  for (int kc = 0; kc < 4; ++kc)
    qbf[kc] = cat8(*(const v8h*)(qrow + kc * 32 + hi * 8),
                   *(const v8h*)(qrow + kc * 32 + hi * 8 + 16));

  float m_i = -__builtin_inff();
  float l_i = 0.f;
  v8f acc[8] = {};   // O accumulators: 8 d-tiles, D-layout (m=t, n=d)
  const int t_glob = qt + r;          // this lane's query row (softmax stats)

  int s0 = 0;
  for (; s0 + 31 <= qt; s0 += 32)     // fully-unmasked key blocks
    attn_block<false>(s0, t_glob, l, r, hi, kb, qbf, vb, m_i, l_i, acc);
  // single diagonal-straddling block (covers keys up to qt+15 per row mask)
  attn_block<true>(s0, t_glob, l, r, hi, kb, qbf, vb, m_i, l_i, acc);

  // ---- epilogue: divide by row sums, store fp32 ----
  float lv[8];
#pragma unroll
  for (int v = 0; v < 8; ++v) lv[v] = bcastf(l_i, v + 8 * hi);
  float* ob = out + (size_t)b * T_DIM * C_DIM;
#pragma unroll
  for (int dt = 0; dt < 8; ++dt) {
    const int d = dt * 16 + r;
#pragma unroll
    for (int v = 0; v < 8; ++v) {
      const int t = qt + v + 8 * hi;
      ob[(size_t)t * C_DIM + d] = acc[dt][v] / lv[v];
    }
  }
}

// ---------------------------------------------------------------------------
extern "C" void kernel_launch(void* const* d_in, const int* in_sizes, int n_in,
                              void* d_out, int out_size, void* d_ws,
                              size_t ws_size, hipStream_t stream) {
  const float* x  = (const float*)d_in[0];
  const float* Wq = (const float*)d_in[1];
  const float* Wk = (const float*)d_in[2];
  const float* Wv = (const float*)d_in[3];
  float* out = (float*)d_out;

  // workspace: q (f16), k (f16), v^T (f16) — 4 MB each — then f16 weights
  _Float16* qh  = (_Float16*)d_ws;
  _Float16* kh  = qh + (size_t)B_DIM * T_DIM * C_DIM;
  _Float16* vth = kh + (size_t)B_DIM * T_DIM * C_DIM;
  _Float16* wf  = vth + (size_t)B_DIM * T_DIM * C_DIM;

  const int nW = 3 * C_DIM * C_DIM;       // 49152 halves, 8 per thread
  cvtw_kernel<<<nW / (256 * 8), 256, 0, stream>>>(Wq, Wk, Wv, wf);

  const int ntiles = B_DIM * T_DIM / 16;  // 1024 row tiles, 4 waves per block
  qkv_kernel<<<ntiles / 4, 128, 0, stream>>>(x, wf, qh, kh, vth);
  attn_kernel<<<ntiles / 4, 128, 0, stream>>>(qh, kh, vth, out);
}
